// Encoder_43911745634974
// MI455X (gfx1250) — matrix operand order repaired and checked by reading, not compiled, not run on hardware
//
#include <hip/hip_runtime.h>
#include <hip/hip_bf16.h>

typedef __bf16 bf16;
typedef __attribute__((ext_vector_type(2)))  __bf16 v2bf;
typedef __attribute__((ext_vector_type(16))) __bf16 v16bf;
typedef __attribute__((ext_vector_type(8)))  float  v8f;

#define D_MODEL 1024
#define D_FF    4096
#define NTOK    4096   // B*S
#define SEQ     1024
#define NHEAD   16
#define DK      64
#define NLAYER  6

// Async global->LDS 16-byte copy (CDNA5 ASYNCcnt path). lds = byte offset
// in LDS (low 32 bits of a flat LDS pointer), g = 64-bit global address.
__device__ __forceinline__ void async_copy_b128(unsigned lds, const void* g) {
    asm volatile("global_load_async_to_lds_b128 %0, %1, off"
                 :: "v"(lds), "v"((unsigned long long)g) : "memory");
}
__device__ __forceinline__ void wait_async0() {
    asm volatile("s_wait_asynccnt 0" ::: "memory");
}
// Pack two f32 -> packed bf16 pair (lowers to one v_cvt_pk_bf16_f32).
__device__ __forceinline__ unsigned pack_bf16(float a, float b) {
    v2bf p;
    p[0] = (bf16)a;
    p[1] = (bf16)b;
    return __builtin_bit_cast(unsigned, p);
}

// ---------------------------------------------------------------------------
// Embedding: x[t,d] = emb[src[t],d]*sqrt(D) + pe[t%S, d]; also bf16 mirror.
// ---------------------------------------------------------------------------
__global__ __launch_bounds__(256) void embed_kernel(
    const int* __restrict__ src, const float* __restrict__ emb,
    const float* __restrict__ pe, float* __restrict__ X, bf16* __restrict__ XB)
{
    int t = blockIdx.x;
    int s = t & (SEQ - 1);
    int tok = src[t];
    int tid = threadIdx.x;
    #pragma unroll
    for (int i = 0; i < 4; ++i) {
        int d = tid + i * 256;
        float v = emb[(size_t)tok * D_MODEL + d] * 32.0f + pe[(size_t)s * D_MODEL + d];
        X [(size_t)t * D_MODEL + d] = v;
        XB[(size_t)t * D_MODEL + d] = (bf16)v;
    }
}

// ---------------------------------------------------------------------------
// WMMA GEMM: C[M,N] = A(bf16)[M,K] @ W(fp32->bf16)[K,N] + bias
// Block: 128 threads (4 waves). Tile 128x64, K-step 32, DOUBLE BUFFERED:
//   A tile staged with global_load_async_to_lds_b128 (no VGPR round-trip),
//   W tile prefetched into registers one K-step ahead; 1 barrier / K-step.
// Wave w computes rows [w*32, w*32+32) x all 64 cols -> 2x4 accumulators.
// mode 0: fp32 row-major   mode 1: bf16 row-major   mode 2: bf16 [B,H,S,Dk]
// ---------------------------------------------------------------------------
#define AS_STR 40
#define BS_STR 40
__global__ __launch_bounds__(128) void gemm_wmma_bf16(
    const bf16* __restrict__ A, const float* __restrict__ W,
    const float* __restrict__ bias,
    float* __restrict__ outF, bf16* __restrict__ outB,
    int M, int K, int N, int mode, int relu)
{
    __shared__ bf16 As[2][128 * AS_STR];  // [row][k], padded
    __shared__ bf16 Bs[2][64 * BS_STR];   // [n][k],  padded (transposed W)

    const int tid  = threadIdx.x;
    const int lane = tid & 31;
    const int wave = tid >> 5;      // 0..3
    const int g    = lane >> 4;     // lane half
    const int mrow = lane & 15;
    const int m0   = blockIdx.y * 128;
    const int n0   = blockIdx.x * 64;
    const int nk   = K >> 5;

    v8f acc[2][4] = {};

    // B staging map: each thread owns column nn and consecutive kk pairs
    const int nn   = tid & 63;
    const int half = tid >> 6;      // 0..1 -> kk range half*16 .. +16
    const float* wcol = W + (size_t)(half * 16) * N + n0 + nn;

    const bf16* arow = A + (size_t)(m0 + tid) * K;
    const unsigned ldsA0 = (unsigned)(size_t)&As[0][tid * AS_STR];
    const unsigned ldsA1 = (unsigned)(size_t)&As[1][tid * AS_STR];

    // ---- prologue: async A(0), W regs(0) ----
    {
        const bf16* sa = arow;
        async_copy_b128(ldsA0,      sa);
        async_copy_b128(ldsA0 + 16, sa + 8);
        async_copy_b128(ldsA0 + 32, sa + 16);
        async_copy_b128(ldsA0 + 48, sa + 24);
    }
    float wr[16];
    {
        const float* wp = wcol;
        #pragma unroll
        for (int j = 0; j < 16; ++j) wr[j] = wp[(size_t)j * N];
    }

    int buf = 0;
    for (int i = 0; i < nk; ++i) {
        // ---- commit W regs -> Bs[buf] (v_cvt_pk_bf16_f32 pairs) ----
        #pragma unroll
        for (int j = 0; j < 8; ++j) {
            int kk = half * 16 + j * 2;
            *(unsigned int*)&Bs[buf][nn * BS_STR + kk] =
                pack_bf16(wr[2 * j], wr[2 * j + 1]);
        }
        wait_async0();          // our async A writes into As[buf] are done
        __syncthreads();        // everyone's staging for tile i visible

        // ---- kick off tile i+1 traffic (overlaps with WMMA below) ----
        if (i + 1 < nk) {
            const bf16* sa = arow + (i + 1) * 32;
            unsigned dl = buf ? ldsA0 : ldsA1;
            async_copy_b128(dl,      sa);
            async_copy_b128(dl + 16, sa + 8);
            async_copy_b128(dl + 32, sa + 16);
            async_copy_b128(dl + 48, sa + 24);
            const float* wp = wcol + (size_t)(i + 1) * 32 * N;
            #pragma unroll
            for (int j = 0; j < 16; ++j) wr[j] = wp[(size_t)j * N];
        }

        // ---- A fragments (2 m-tiles per wave) ----
        v16bf a[2];
        #pragma unroll
        for (int mt = 0; mt < 2; ++mt) {
            const bf16* p0 = &As[buf][(wave * 32 + mt * 16 + mrow) * AS_STR + g * 16];
            const bf16* p1 = p0 + 8;
            #pragma unroll
            for (int ii = 0; ii < 8; ++ii) { a[mt][ii] = p0[ii]; a[mt][8 + ii] = p1[ii]; }
        }
        // ---- 4 B fragments, each reused by 2 m-tiles ----
        #pragma unroll
        for (int nt = 0; nt < 4; ++nt) {
            const bf16* pb = &Bs[buf][(nt * 16 + mrow) * BS_STR + g * 16];
            v16bf b;
            #pragma unroll
            for (int ii = 0; ii < 16; ++ii) b[ii] = pb[ii];
            #pragma unroll
            for (int mt = 0; mt < 2; ++mt)
                acc[mt][nt] = __builtin_amdgcn_wmma_f32_16x16x32_bf16(
                    false, a[mt], false, b, (short)0, acc[mt][nt], false, false);
        }
        buf ^= 1;
    }

    // ---- epilogue ----
    #pragma unroll
    for (int mt = 0; mt < 2; ++mt)
        #pragma unroll
        for (int nt = 0; nt < 4; ++nt) {
            int col = n0 + nt * 16 + mrow;
            float bv = bias[col];
            #pragma unroll
            for (int r = 0; r < 8; ++r) {
                int row = m0 + wave * 32 + mt * 16 + r + 8 * g;
                float v = acc[mt][nt][r] + bv;
                if (relu) v = v > 0.0f ? v : 0.0f;
                if (mode == 0) {
                    outF[(size_t)row * N + col] = v;
                } else if (mode == 1) {
                    outB[(size_t)row * N + col] = (bf16)v;
                } else { // [B,H,S,Dk] scatter for Q/K/V
                    int b  = row >> 10, s = row & (SEQ - 1);
                    int h  = col >> 6,  dk = col & 63;
                    outB[(((size_t)(b * NHEAD + h)) * SEQ + s) * DK + dk] = (bf16)v;
                }
            }
        }
}

// ---------------------------------------------------------------------------
// Flash attention. Block = 128 threads (4 waves); each wave owns one 16-row
// q tile; the 4 waves SHARE the staged K/V chunks. K chunk staged via async
// global->LDS; online softmax; WMMA for QK^T and P*V.
// ---------------------------------------------------------------------------
#define KT_STR 72
#define VT_STR 72
#define PT_STR 72
__global__ __launch_bounds__(128) void attn_kernel(
    const bf16* __restrict__ Q, const bf16* __restrict__ K,
    const bf16* __restrict__ V, const int* __restrict__ mask,
    bf16* __restrict__ out)
{
    __shared__ bf16 Kt[64 * KT_STR];      // [kv][dk]  (WMMA B layout for Q*K^T)
    __shared__ bf16 Vt[64 * VT_STR];      // [dk][kv]  (WMMA B layout for P*V)
    __shared__ bf16 Pt[4][16 * PT_STR];   // per-wave probs staging

    const int tid  = threadIdx.x;
    const int lane = tid & 31;
    const int wave = tid >> 5;
    const int g    = lane >> 4;
    const int mrow = lane & 15;
    const int bid  = blockIdx.x;          // B*H*(S/64) = 1024 blocks
    const int qt   = (bid & 15) * 4 + wave;
    const int h    = (bid >> 4) & (NHEAD - 1);
    const int b    = bid >> 8;
    const size_t headOff = ((size_t)(b * NHEAD + h)) * SEQ * DK;

    // Q A-fragments for both dk k-steps, straight from global
    v16bf qa[2];
    {
        const bf16* qp = Q + headOff + (size_t)(qt * 16 + mrow) * DK;
        #pragma unroll
        for (int ks = 0; ks < 2; ++ks)
            #pragma unroll
            for (int i = 0; i < 8; ++i) {
                qa[ks][i]     = qp[ks * 32 + g * 16 + i];
                qa[ks][8 + i] = qp[ks * 32 + g * 16 + 8 + i];
            }
    }

    float m[8], l[8];
    #pragma unroll
    for (int r = 0; r < 8; ++r) { m[r] = -1e30f; l[r] = 0.0f; }
    v8f o[4] = {};

    for (int c = 0; c < SEQ / 64; ++c) {
        const int kv0 = c * 64;
        // stage K chunk via async copy: 512 16B chunks, 4 per thread
        #pragma unroll
        for (int it = 0; it < 4; ++it) {
            int chunk = it * 128 + tid;
            int row = chunk >> 3, coff = (chunk & 7) * 8;
            async_copy_b128((unsigned)(size_t)&Kt[row * KT_STR + coff],
                            K + headOff + (size_t)(kv0 + row) * DK + coff);
        }
        // stage V chunk transposed, kv-pair packed: Vt[d][2p..2p+1] <- b32
        {
            int p  = tid & 31;            // kv pair index
            int d0 = (tid >> 5) * 16;     // 16 d values per thread
            const unsigned short* v0 =
                (const unsigned short*)(V + headOff + (size_t)(kv0 + 2 * p) * DK);
            const unsigned short* v1 = v0 + DK;
            unsigned short r0[16], r1[16];
            #pragma unroll
            for (int j = 0; j < 16; ++j) { r0[j] = v0[d0 + j]; r1[j] = v1[d0 + j]; }
            unsigned int* dst = (unsigned int*)Vt;
            #pragma unroll
            for (int j = 0; j < 16; ++j)
                dst[(d0 + j) * (VT_STR / 2) + p] =
                    (unsigned int)r0[j] | ((unsigned int)r1[j] << 16);
        }
        wait_async0();
        __syncthreads();

        // scores = Q @ K^T (4 kv-tiles x 2 dk-steps)
        v8f s[4] = {};
        #pragma unroll
        for (int nt = 0; nt < 4; ++nt)
            #pragma unroll
            for (int ks = 0; ks < 2; ++ks) {
                const bf16* pb = &Kt[(nt * 16 + mrow) * KT_STR + ks * 32 + g * 16];
                v16bf bfr;
                #pragma unroll
                for (int i = 0; i < 16; ++i) bfr[i] = pb[i];
                s[nt] = __builtin_amdgcn_wmma_f32_16x16x32_bf16(
                    false, qa[ks], false, bfr, (short)0, s[nt], false, false);
            }

        // scale + mask + chunk row-max
        float mc[8];
        #pragma unroll
        for (int r = 0; r < 8; ++r) mc[r] = -1e30f;
        #pragma unroll
        for (int nt = 0; nt < 4; ++nt) {
            float madd = (mask[b * SEQ + kv0 + nt * 16 + mrow] == 0) ? -1e9f : 0.0f;
            #pragma unroll
            for (int r = 0; r < 8; ++r) {
                float v = s[nt][r] * 0.125f + madd;   // Dk^-0.5 = 1/8
                s[nt][r] = v;
                mc[r] = fmaxf(mc[r], v);
            }
        }
        // reduce row-max across the 16 lanes of each half
        #pragma unroll
        for (int off = 1; off < 16; off <<= 1)
            #pragma unroll
            for (int r = 0; r < 8; ++r)
                mc[r] = fmaxf(mc[r], __shfl_xor(mc[r], off, 32));

        float f[8], ls[8];
        #pragma unroll
        for (int r = 0; r < 8; ++r) {
            float mn = fmaxf(m[r], mc[r]);
            f[r] = __expf(m[r] - mn);
            m[r] = mn;
            ls[r] = 0.0f;
        }
        #pragma unroll
        for (int nt = 0; nt < 4; ++nt)
            #pragma unroll
            for (int r = 0; r < 8; ++r) {
                float p = __expf(s[nt][r] - m[r]);
                s[nt][r] = p;
                ls[r] += p;
            }
        #pragma unroll
        for (int off = 1; off < 16; off <<= 1)
            #pragma unroll
            for (int r = 0; r < 8; ++r)
                ls[r] += __shfl_xor(ls[r], off, 32);
        #pragma unroll
        for (int r = 0; r < 8; ++r) l[r] = l[r] * f[r] + ls[r];

        // stage P (C-layout -> row-major LDS, per-wave slice)
        #pragma unroll
        for (int nt = 0; nt < 4; ++nt)
            #pragma unroll
            for (int r = 0; r < 8; ++r)
                Pt[wave][(r + 8 * g) * PT_STR + nt * 16 + mrow] = (bf16)s[nt][r];
        __syncthreads();

        // rescale O, then O += P @ V
        #pragma unroll
        for (int nt = 0; nt < 4; ++nt)
            #pragma unroll
            for (int r = 0; r < 8; ++r) o[nt][r] *= f[r];

        v16bf pa[2];
        #pragma unroll
        for (int ks = 0; ks < 2; ++ks)
            #pragma unroll
            for (int i = 0; i < 8; ++i) {
                pa[ks][i]     = Pt[wave][mrow * PT_STR + ks * 32 + g * 16 + i];
                pa[ks][8 + i] = Pt[wave][mrow * PT_STR + ks * 32 + g * 16 + 8 + i];
            }
        #pragma unroll
        for (int nt = 0; nt < 4; ++nt)
            #pragma unroll
            for (int ks = 0; ks < 2; ++ks) {
                const bf16* pb = &Vt[(nt * 16 + mrow) * VT_STR + ks * 32 + g * 16];
                v16bf bfr;
                #pragma unroll
                for (int i = 0; i < 16; ++i) bfr[i] = pb[i];
                o[nt] = __builtin_amdgcn_wmma_f32_16x16x32_bf16(
                    false, pa[ks], false, bfr, (short)0, o[nt], false, false);
            }
        __syncthreads();
    }

    // write attention output bf16 [token, D] for O-projection
    #pragma unroll
    for (int nt = 0; nt < 4; ++nt)
        #pragma unroll
        for (int r = 0; r < 8; ++r) {
            int row = b * SEQ + qt * 16 + r + 8 * g;
            int col = h * DK + nt * 16 + mrow;
            out[(size_t)row * D_MODEL + col] = (bf16)(o[nt][r] / l[r]);
        }
}

// ---------------------------------------------------------------------------
// Residual + LayerNorm: y = LN(x + a)*g + b; writes fp32 + bf16 mirrors.
// One 256-thread block per row of 1024.
// ---------------------------------------------------------------------------
__global__ __launch_bounds__(256) void ln_residual_kernel(
    const float* __restrict__ x, const float* __restrict__ a,
    const float* __restrict__ gamma, const float* __restrict__ beta,
    float* __restrict__ outF, bf16* __restrict__ outB)
{
    __shared__ float ps[8], ps2[8];
    const int row = blockIdx.x;
    const int tid = threadIdx.x;
    const float4 xv = ((const float4*)(x + (size_t)row * D_MODEL))[tid];
    const float4 av = ((const float4*)(a + (size_t)row * D_MODEL))[tid];
    float v0 = xv.x + av.x, v1 = xv.y + av.y, v2 = xv.z + av.z, v3 = xv.w + av.w;
    float s  = v0 + v1 + v2 + v3;
    float s2 = v0 * v0 + v1 * v1 + v2 * v2 + v3 * v3;
    #pragma unroll
    for (int off = 1; off < 32; off <<= 1) {
        s  += __shfl_xor(s,  off, 32);
        s2 += __shfl_xor(s2, off, 32);
    }
    int wv = tid >> 5, ln = tid & 31;
    if (ln == 0) { ps[wv] = s; ps2[wv] = s2; }
    __syncthreads();
    if (wv == 0) {
        float t  = ln < 8 ? ps[ln]  : 0.0f;
        float t2 = ln < 8 ? ps2[ln] : 0.0f;
        #pragma unroll
        for (int off = 1; off < 8; off <<= 1) {
            t  += __shfl_xor(t,  off, 32);
            t2 += __shfl_xor(t2, off, 32);
        }
        if (ln == 0) { ps[0] = t; ps2[0] = t2; }
    }
    __syncthreads();
    const float mu  = ps[0] * (1.0f / D_MODEL);
    const float var = ps2[0] * (1.0f / D_MODEL) - mu * mu;
    const float rstd = rsqrtf(var + 1e-5f);
    float vv[4] = { v0, v1, v2, v3 };
    #pragma unroll
    for (int i = 0; i < 4; ++i) {
        int d = tid * 4 + i;
        float y = (vv[i] - mu) * rstd * gamma[d] + beta[d];
        outF[(size_t)row * D_MODEL + d] = y;
        outB[(size_t)row * D_MODEL + d] = (bf16)y;
    }
}

// ---------------------------------------------------------------------------
extern "C" void kernel_launch(void* const* d_in, const int* in_sizes, int n_in,
                              void* d_out, int out_size, void* d_ws, size_t ws_size,
                              hipStream_t stream) {
    const int*   src  = (const int*)  d_in[0];
    const int*   mask = (const int*)  d_in[1];
    const float* emb  = (const float*)d_in[2];
    const float* pe   = (const float*)d_in[3];
    const float* wq   = (const float*)d_in[4];
    const float* bq   = (const float*)d_in[5];
    const float* wk   = (const float*)d_in[6];
    const float* bk   = (const float*)d_in[7];
    const float* wv   = (const float*)d_in[8];
    const float* bv   = (const float*)d_in[9];
    const float* wo   = (const float*)d_in[10];
    const float* bo   = (const float*)d_in[11];
    const float* w1   = (const float*)d_in[12];
    const float* b1   = (const float*)d_in[13];
    const float* w2   = (const float*)d_in[14];
    const float* b2   = (const float*)d_in[15];
    const float* g1   = (const float*)d_in[16];
    const float* be1  = (const float*)d_in[17];
    const float* g2   = (const float*)d_in[18];
    const float* be2  = (const float*)d_in[19];
    float* out = (float*)d_out;

    char* ws = (char*)d_ws;
    const size_t MB = 1ull << 20;
    float* X   = (float*)(ws);             // 16 MB  fp32 residual stream
    bf16*  XB  = (bf16*)(ws + 16 * MB);    //  8 MB  bf16 mirror
    float* T   = (float*)(ws + 24 * MB);   // 16 MB  fp32 GEMM out
    bf16*  QB  = (bf16*)(ws + 40 * MB);    //  8 MB  [B,H,S,Dk]
    bf16*  KB  = (bf16*)(ws + 48 * MB);
    bf16*  VB  = (bf16*)(ws + 56 * MB);
    bf16*  ATB = (bf16*)(ws + 64 * MB);    //  8 MB  attention out
    bf16*  HB  = (bf16*)(ws + 72 * MB);    // 32 MB  FFN hidden

    embed_kernel<<<NTOK, 256, 0, stream>>>(src, emb, pe, X, XB);

    const dim3 gProj(D_MODEL / 64, NTOK / 128);  // 4096x1024x1024
    const dim3 gFF1 (D_FF    / 64, NTOK / 128);  // 4096x1024x4096
    const dim3 gFF2 (D_MODEL / 64, NTOK / 128);  // 4096x4096x1024

    for (int l = 0; l < NLAYER; ++l) {
        const size_t wOff = (size_t)l * D_MODEL * D_MODEL;
        const size_t fOff = (size_t)l * D_MODEL * D_FF;
        gemm_wmma_bf16<<<gProj, 128, 0, stream>>>(XB, wq + wOff, bq + l * D_MODEL,
            nullptr, QB, NTOK, D_MODEL, D_MODEL, 2, 0);
        gemm_wmma_bf16<<<gProj, 128, 0, stream>>>(XB, wk + wOff, bk + l * D_MODEL,
            nullptr, KB, NTOK, D_MODEL, D_MODEL, 2, 0);
        gemm_wmma_bf16<<<gProj, 128, 0, stream>>>(XB, wv + wOff, bv + l * D_MODEL,
            nullptr, VB, NTOK, D_MODEL, D_MODEL, 2, 0);

        attn_kernel<<<NTOK / 4, 128, 0, stream>>>(QB, KB, VB, mask, ATB);

        gemm_wmma_bf16<<<gProj, 128, 0, stream>>>(ATB, wo + wOff, bo + l * D_MODEL,
            T, nullptr, NTOK, D_MODEL, D_MODEL, 0, 0);
        ln_residual_kernel<<<NTOK, 256, 0, stream>>>(X, T,
            g1 + l * D_MODEL, be1 + l * D_MODEL, X, XB);

        gemm_wmma_bf16<<<gFF1, 128, 0, stream>>>(XB, w1 + fOff, b1 + l * D_FF,
            nullptr, HB, NTOK, D_MODEL, D_FF, 1, 1);
        gemm_wmma_bf16<<<gFF2, 128, 0, stream>>>(HB, w2 + fOff, b2 + l * D_MODEL,
            T, nullptr, NTOK, D_FF, D_MODEL, 0, 0);

        float* lnout = (l == NLAYER - 1) ? out : X;
        ln_residual_kernel<<<NTOK, 256, 0, stream>>>(X, T,
            g2 + l * D_MODEL, be2 + l * D_MODEL, lnout, XB);
    }
}
